// Model_87050397155809
// MI455X (gfx1250) — matrix operand order repaired and checked by reading, not compiled, not run on hardware
//
#include <hip/hip_runtime.h>

// ---------------------------------------------------------------------------
// CDNA5 (gfx1250) implicit-GEMM conv3x3 encoder using v_wmma_f32_16x16x32_bf16
// A-fragments built with ds_load_b128 via window-major K ordering.
// Native bf16 converts (v_cvt_*_bf16) for all f32<->bf16 traffic.
// ---------------------------------------------------------------------------

typedef __bf16         v16bf __attribute__((ext_vector_type(16)));
typedef __bf16         v2bf  __attribute__((ext_vector_type(2)));
typedef float          v8f   __attribute__((ext_vector_type(8)));
typedef unsigned short v16u  __attribute__((ext_vector_type(16)));
typedef unsigned int   v8ui  __attribute__((ext_vector_type(8)));
typedef unsigned int   v4ui  __attribute__((ext_vector_type(4)));

__device__ __forceinline__ unsigned short f2bf(float f) {
  __bf16 b = (__bf16)f;                       // native cvt, RNE
  return __builtin_bit_cast(unsigned short, b);
}
__device__ __forceinline__ unsigned int pack2bf(float f0, float f1) {
  v2bf p;
  p[0] = (__bf16)f0;                          // should select v_cvt_pk_bf16_f32
  p[1] = (__bf16)f1;
  return __builtin_bit_cast(unsigned int, p);
}
__device__ __forceinline__ float bf2f(unsigned short h) {
  return (float)__builtin_bit_cast(__bf16, h);
}
__device__ __forceinline__ float leaky(float v) { return v > 0.f ? v : 0.1f * v; }

// ---------------------------------------------------------------------------
// Pack OIHW fp32 weights into the bf16 B-fragment lane layout:
//   pk[((kc*3 + t)*32 + lane)*16 + e] = B[K][N],
//   K = kc*32 + (e<8 ? e : e+8) + 8*(lane>=16), N = t*16 + (lane&15)
// K ordering: order==0: k = cin*9 + (ky*3+kx)        (scalar-gather path, Cin=3)
//             order==1: k = (ky*3+kx)*CIN + cin      (ds_b128 fast path, Cin=48)
// ---------------------------------------------------------------------------
__global__ void pack_w(const float* __restrict__ w, unsigned short* __restrict__ pk,
                       int CIN, int KC, int order) {
  int idx = blockIdx.x * 256 + threadIdx.x;
  int total = KC * 1536;                   // 3 tiles * 32 lanes * 16 elems / chunk
  if (idx >= total) return;
  int e  = idx & 15;
  int L  = (idx >> 4) & 31;
  int t  = (idx >> 9) % 3;
  int kc = idx / 1536;
  int K  = (e < 8 ? e : e + 8) + ((L >> 4) << 3);
  int k  = kc * 32 + K;
  int n  = t * 16 + (L & 15);
  unsigned short v = 0;
  int Ktot = CIN * 9;
  if (k < Ktot) {
    int cin, r;
    if (order == 0) { cin = k / 9;   r = k - cin * 9; }
    else            { r   = k / CIN; cin = k - r * CIN; }
    v = f2bf(w[(n * CIN + cin) * 9 + r]);
  }
  pk[idx] = v;
}

// ---------------------------------------------------------------------------
// conv3x3, stride 1, pad 1, bias + LeakyReLU(0.1), bf16 out.
// Block = 256 threads = 8 waves; block tile = 16(x) x 8(y) output pixels,
// each wave owns one row of 16 pixels and all 48 output channels.
// ---------------------------------------------------------------------------
template <int CIN, int KC, bool IN_FLOAT>
__global__ __launch_bounds__(256) void conv3x3_wmma(
    const void* __restrict__ in_, const unsigned short* __restrict__ pkW,
    const float* __restrict__ bias, unsigned short* __restrict__ out,
    int H, int W) {
  constexpr bool FASTK = (CIN % 8 == 0);
  constexpr int  CPAD  = 56;  // cin padding: 112B spatial stride, 16B aligned
  __shared__ unsigned short lds[FASTK ? 180 * CPAD : CIN * 180];

  const int tid = threadIdx.x;
  const int n   = blockIdx.z;
  const int x0  = blockIdx.x * 16;
  const int y0  = blockIdx.y * 8;

  // ---- cooperative stage of input tile (with 1-px halo) into LDS as bf16 ----
  const float*          inf = (const float*)in_;
  const unsigned short* inu = (const unsigned short*)in_;
  for (int i = tid; i < CIN * 180; i += 256) {
    int cin = i / 180;              // cin-major loop keeps global reads coalesced
    int rem = i - cin * 180;
    int dy  = rem / 18;
    int dx  = rem - dy * 18;
    int gy  = y0 - 1 + dy;
    int gx  = x0 - 1 + dx;
    unsigned short v = 0;
    if (gy >= 0 && gy < H && gx >= 0 && gx < W) {
      int gi = ((n * CIN + cin) * H + gy) * W + gx;
      v = IN_FLOAT ? f2bf(inf[gi]) : inu[gi];
    }
    lds[FASTK ? (rem * CPAD + cin) : (cin * 180 + rem)] = v;
  }
  __syncthreads();

  const int lane = tid & 31;
  const int wy   = tid >> 5;      // wave = output row in tile
  const int m    = lane & 15;     // A-fragment M (= pixel x offset)
  const int hi   = lane >> 4;

  const v8ui* bglob = (const v8ui*)pkW;  // 32B per lane-fragment
  v8f acc0 = {}, acc1 = {}, acc2 = {};
  const int Ktot = CIN * 9;

  for (int kc = 0; kc < KC; ++kc) {
    v16bf A;
    if constexpr (FASTK) {
      // K-order k = (ky*3+kx)*CIN + cin, LDS [spatial][cin]: each 8-elem run of
      // the A fragment is 16 contiguous bytes in LDS -> one ds_load_b128.
      auto loadrun = [&](int k) -> v4ui {
        v4ui r = {0u, 0u, 0u, 0u};
        if (k < Ktot) {
          int wdw = k / CIN;                 // 3x3 window index (0..8)
          int c   = k - wdw * CIN;           // cin base (multiple of 8)
          int ky  = wdw / 3;
          int kx  = wdw - ky * 3;
          r = *(const v4ui*)&lds[((wy + ky) * 18 + (m + kx)) * CPAD + c];
        }
        return r;
      };
      v4ui r0 = loadrun(kc * 32 + (hi << 3));        // e = 0..7
      v4ui r1 = loadrun(kc * 32 + 16 + (hi << 3));   // e = 8..15
      v8ui a8 = __builtin_shufflevector(r0, r1, 0, 1, 2, 3, 4, 5, 6, 7);
      A = __builtin_bit_cast(v16bf, a8);
    } else {
      // scalar gather path (Cin=3 only): K-order k = cin*9 + ky*3 + kx
      v16u au;
#pragma unroll
      for (int e = 0; e < 16; ++e) {
        int K = (e < 8 ? e : e + 8) + (hi << 3);
        int k = kc * 32 + K;
        unsigned short v = 0;
        if (k < Ktot) {
          int cin = k / 9;
          int r   = k - cin * 9;
          int ky  = r / 3;
          int kx  = r - ky * 3;
          v = lds[cin * 180 + (wy + ky) * 18 + (m + kx)];
        }
        au[e] = v;
      }
      A = __builtin_bit_cast(v16bf, au);
    }

    if (kc + 1 < KC)  // prefetch next chunk's packed weights (global_prefetch_b8)
      __builtin_prefetch((const void*)(bglob + (kc + 1) * 96 + lane), 0, 3);

    v16bf B0 = __builtin_bit_cast(v16bf, bglob[(kc * 3 + 0) * 32 + lane]);
    v16bf B1 = __builtin_bit_cast(v16bf, bglob[(kc * 3 + 1) * 32 + lane]);
    v16bf B2 = __builtin_bit_cast(v16bf, bglob[(kc * 3 + 2) * 32 + lane]);

    acc0 = __builtin_amdgcn_wmma_f32_16x16x32_bf16(false, A, false, B0, (short)0, acc0, false, false);
    acc1 = __builtin_amdgcn_wmma_f32_16x16x32_bf16(false, A, false, B1, (short)0, acc1, false, false);
    acc2 = __builtin_amdgcn_wmma_f32_16x16x32_bf16(false, A, false, B2, (short)0, acc2, false, false);
  }

  // ---- epilogue: bias + LeakyReLU, pack 8 bf16 -> one 16B store per tile ----
  // D layout: element r of acc, lane L -> M = r + 8*(L>=16), N = L&15
  const int y  = y0 + wy;
  const int xb = x0 + (hi << 3);
  v8f accs[3] = {acc0, acc1, acc2};
#pragma unroll
  for (int t = 0; t < 3; ++t) {
    int cout = t * 16 + (lane & 15);
    float bb = bias[cout];
    unsigned int dw[4];
#pragma unroll
    for (int r = 0; r < 4; ++r) {
      float f0 = leaky(accs[t][2 * r] + bb);
      float f1 = leaky(accs[t][2 * r + 1] + bb);
      dw[r] = pack2bf(f0, f1);                       // packed bf16 cvt
    }
    uint4 pkd = {dw[0], dw[1], dw[2], dw[3]};
    int base = ((n * 48 + cout) * H + y) * W + xb;   // 32-bit addressing: <2^31
    *(uint4*)(out + base) = pkd;
  }
}

// ---------------------------------------------------------------------------
// 2x2 max-pool, bf16 in; bf16 or fp32 out (final layer writes fp32 to d_out)
// ---------------------------------------------------------------------------
template <bool OUT_FLOAT>
__global__ void maxpool2(const unsigned short* __restrict__ in, void* __restrict__ out,
                         int Hi, int Wi, int total) {
  int idx = blockIdx.x * 256 + threadIdx.x;
  if (idx >= total) return;
  int Wo = Wi >> 1, Ho = Hi >> 1;
  int x  = idx % Wo;
  int y  = (idx / Wo) % Ho;
  int nc = idx / (Wo * Ho);
  int b  = (nc * Hi + 2 * y) * Wi + 2 * x;
  float mx = fmaxf(fmaxf(bf2f(in[b]), bf2f(in[b + 1])),
                   fmaxf(bf2f(in[b + Wi]), bf2f(in[b + Wi + 1])));
  if (OUT_FLOAT) ((float*)out)[idx] = mx;
  else           ((unsigned short*)out)[idx] = f2bf(mx);
}

// ---------------------------------------------------------------------------
extern "C" void kernel_launch(void* const* d_in, const int* in_sizes, int n_in,
                              void* d_out, int out_size, void* d_ws, size_t ws_size,
                              hipStream_t stream) {
  const float* x  = (const float*)d_in[0];
  const float* w1 = (const float*)d_in[1];
  const float* b1 = (const float*)d_in[2];
  const float* w2 = (const float*)d_in[3];
  const float* b2 = (const float*)d_in[4];
  const float* w3 = (const float*)d_in[5];
  const float* b3 = (const float*)d_in[6];
  const float* w4 = (const float*)d_in[7];
  const float* b4 = (const float*)d_in[8];

  char* ws = (char*)d_ws;
  size_t off = 0;
  auto alloc = [&](size_t bytes) {
    void* p = ws + off;
    off = (off + bytes + 255) & ~(size_t)255;
    return p;
  };
  unsigned short* pk1 = (unsigned short*)alloc((size_t)1 * 1536 * 2);
  unsigned short* pk2 = (unsigned short*)alloc((size_t)14 * 1536 * 2);
  unsigned short* pk3 = (unsigned short*)alloc((size_t)14 * 1536 * 2);
  unsigned short* pk4 = (unsigned short*)alloc((size_t)14 * 1536 * 2);
  const size_t actBytes = (size_t)8 * 48 * 256 * 256 * 2;  // 50.3 MB bf16
  unsigned short* bufA = (unsigned short*)alloc(actBytes);
  unsigned short* bufB = (unsigned short*)alloc(actBytes);

  // pack weights into WMMA B-fragment layout (re-done every call: deterministic)
  pack_w<<<(1 * 1536 + 255) / 256, 256, 0, stream>>>(w1, pk1, 3, 1, 0);
  pack_w<<<(14 * 1536 + 255) / 256, 256, 0, stream>>>(w2, pk2, 48, 14, 1);
  pack_w<<<(14 * 1536 + 255) / 256, 256, 0, stream>>>(w3, pk3, 48, 14, 1);
  pack_w<<<(14 * 1536 + 255) / 256, 256, 0, stream>>>(w4, pk4, 48, 14, 1);

  // conv1: 3->48 @256^2 (fp32 in), conv2: 48->48 @256^2, pool -> 128^2
  conv3x3_wmma<3, 1, true><<<dim3(16, 32, 8), 256, 0, stream>>>(x, pk1, b1, bufA, 256, 256);
  conv3x3_wmma<48, 14, false><<<dim3(16, 32, 8), 256, 0, stream>>>(bufA, pk2, b2, bufB, 256, 256);
  int tp2 = 8 * 48 * 128 * 128;
  maxpool2<false><<<(tp2 + 255) / 256, 256, 0, stream>>>(bufB, bufA, 256, 256, tp2);

  // conv3 @128^2, pool -> 64^2
  conv3x3_wmma<48, 14, false><<<dim3(8, 16, 8), 256, 0, stream>>>(bufA, pk3, b3, bufB, 128, 128);
  int tp3 = 8 * 48 * 64 * 64;
  maxpool2<false><<<(tp3 + 255) / 256, 256, 0, stream>>>(bufB, bufA, 128, 128, tp3);

  // conv4 @64^2, final pool -> 32^2 fp32 to d_out
  conv3x3_wmma<48, 14, false><<<dim3(4, 8, 8), 256, 0, stream>>>(bufA, pk4, b4, bufB, 64, 64);
  int tp4 = 8 * 48 * 32 * 32;
  maxpool2<true><<<(tp4 + 255) / 256, 256, 0, stream>>>(bufB, d_out, 64, 64, tp4);

  (void)in_sizes; (void)n_in; (void)out_size; (void)ws_size;
}
